// GraphComparator_32057635897399
// MI455X (gfx1250) — compile-verified, matrix operand order checked by reading, not run on hardware
//
#include <hip/hip_runtime.h>
#include <hip/hip_bf16.h>

typedef __attribute__((ext_vector_type(16))) _Float16 v16h;
typedef __attribute__((ext_vector_type(8)))  _Float16 v8h;
typedef __attribute__((ext_vector_type(8)))  float    v8f;

#define D_DIM 384
#define L_DIM 512
#define B_DIM 64
#define N_ROWS (B_DIM * L_DIM)   // 32768

// ---------------------------------------------------------------- utilities

__global__ __launch_bounds__(256)
void zero_kernel(float* __restrict__ p, int n) {
    int i = blockIdx.x * 256 + threadIdx.x;
    if (i < n) p[i] = 0.0f;
}

__global__ __launch_bounds__(256)
void count_kernel(const int* __restrict__ bo, const int* __restrict__ bm,
                  int* __restrict__ cnt_o, int* __restrict__ cnt_m, int n) {
    int i = blockIdx.x * 256 + threadIdx.x;
    if (i < n) {
        atomicAdd(&cnt_o[bo[i]], 1);
        atomicAdd(&cnt_m[bm[i]], 1);
    }
}

__global__ void offsets_kernel(const int* __restrict__ cnt_o, const int* __restrict__ cnt_m,
                               int* __restrict__ off_o, int* __restrict__ off_m) {
    if (threadIdx.x == 0 && blockIdx.x == 0) {
        int a = 0, c = 0;
        for (int i = 0; i < B_DIM; ++i) {
            off_o[i] = a; a += cnt_o[i];
            off_m[i] = c; c += cnt_m[i];
        }
    }
}

// -------------------------------------- CDNA5 async global->LDS copy (16 B)

__device__ __forceinline__ void async_ld16(uint32_t lds_off, const void* gsrc) {
    // GLOBAL_LOAD_ASYNC_TO_LDS_B128: copies 16 B from global memory into LDS, tracked by ASYNCcnt
    asm volatile("global_load_async_to_lds_b128 %0, %1, off"
                 :: "v"(lds_off), "v"((uint64_t)(uintptr_t)gsrc)
                 : "memory");
}

__device__ __forceinline__ void wait_async0() {
    asm volatile("s_wait_asynccnt 0" ::: "memory");
}

// low 32 bits of a generic pointer to __shared__ = wave-relative LDS byte offset
__device__ __forceinline__ uint32_t lds_off32(const void* p) {
    return (uint32_t)(uintptr_t)p;
}

// ------------------------------------------------- gather + LayerNorm (f16)

__device__ __forceinline__ float block_sum_128(float v, float* red, int t) {
#pragma unroll
    for (int off = 1; off < 32; off <<= 1) v += __shfl_xor(v, off, 32);
    __syncthreads();
    if ((t & 31) == 0) red[t >> 5] = v;
    __syncthreads();
    return red[0] + red[1] + red[2] + red[3];
}

__global__ __launch_bounds__(128)
void gather_ln_kernel(const float* __restrict__ h_orig, const float* __restrict__ h_mut,
                      const float* __restrict__ gamma,  const float* __restrict__ beta,
                      const int* __restrict__ alignment, const int* __restrict__ source,
                      const int* __restrict__ cnt_o, const int* __restrict__ cnt_m,
                      const int* __restrict__ off_o, const int* __restrict__ off_m,
                      _Float16* __restrict__ aO, _Float16* __restrict__ aM) {
    __shared__ float red[4];
    const int t   = threadIdx.x;
    const int pos = blockIdx.x;            // b*512 + l
    const int b   = pos >> 9;
    const int l   = pos & 511;

    const int idx    = alignment[pos];
    const int is_mut = source[pos];
    const int lo_    = cnt_o[b];
    const int lm_    = cnt_m[b];
    const bool valid = (idx >= 0) &&
        (is_mut ? (idx < lo_ && l < lm_) : (idx < lm_ && l < lo_));

    const size_t outoff = (size_t)pos * D_DIM;
    if (!valid) {
#pragma unroll
        for (int j = 0; j < 3; ++j) {
            aO[outoff + t + j * 128] = (_Float16)0.0f;
            aM[outoff + t + j * 128] = (_Float16)0.0f;
        }
        return;
    }
    int ol = is_mut ? idx : l;
    int ml = is_mut ? l : idx;
    ol = min(max(ol, 0), lo_ - 1);
    ml = min(max(ml, 0), lm_ - 1);
    const float* srcs[2] = { h_orig + (size_t)(off_o[b] + ol) * D_DIM,
                             h_mut  + (size_t)(off_m[b] + ml) * D_DIM };
    _Float16* dsts[2] = { aO + outoff, aM + outoff };

    const float g0 = gamma[t], g1 = gamma[t + 128], g2 = gamma[t + 256];
    const float e0 = beta[t],  e1 = beta[t + 128],  e2 = beta[t + 256];

#pragma unroll
    for (int r = 0; r < 2; ++r) {
        const float* src = srcs[r];
        float x0 = src[t], x1 = src[t + 128], x2 = src[t + 256];
        float mu = block_sum_128(x0 + x1 + x2, red, t) * (1.0f / 384.0f);
        float d0 = x0 - mu, d1 = x1 - mu, d2 = x2 - mu;
        float var = block_sum_128(d0 * d0 + d1 * d1 + d2 * d2, red, t) * (1.0f / 384.0f);
        float inv = rsqrtf(var + 1e-5f);
        _Float16* dst = dsts[r];
        dst[t]       = (_Float16)(d0 * inv * g0 + e0);
        dst[t + 128] = (_Float16)(d1 * inv * g1 + e1);
        dst[t + 256] = (_Float16)(d2 * inv * g2 + e2);
    }
}

// ----------------------------------------------- WMMA attention (scores + w)

__device__ __forceinline__ v16h ld_frag(const _Float16* p) {
    // 16-bit A/B fragment: halves 0..7 = K[kh..kh+7], halves 8..15 = K[kh+16..kh+23]
    v8h lo = *(const v8h*)(p);
    v8h hi = *(const v8h*)(p + 16);
    v16h r;
#pragma unroll
    for (int i = 0; i < 8; ++i) { r[i] = lo[i]; r[i + 8] = hi[i]; }
    return r;
}

__global__ __launch_bounds__(256)
void attn_kernel(const _Float16* __restrict__ aO,
                 const _Float16* __restrict__ aM,
                 float* __restrict__ wcol) {
    constexpr int MST = 40;                        // padded LDS row stride (halves)
    __shared__ _Float16 Mb[2][L_DIM * MST];        // 2 x 40 KB (double buffer)
    __shared__ _Float16 Ob[2][32 * MST];           // 2 x 2.5 KB
    __shared__ float red[8 * 32];
    __shared__ float rowmax[32];
    __shared__ float rowinv[32];

    const int tid  = threadIdx.x;
    const int wv   = tid >> 5;                     // wave 0..7 -> 64-col strip
    const int lane = tid & 31;
    const int n    = lane & 15;
    const int half = lane >> 4;
    const int kh   = half * 8;                     // K-half offset within fragment
    const int mb8  = half * 8;                     // row-half offset within D tile
    const int b    = blockIdx.y;
    const int l0   = blockIdx.x * 32;
    const size_t bbase = (size_t)b * L_DIM * D_DIM;

    const uint32_t mb_lds[2] = { lds_off32(Mb[0]), lds_off32(Mb[1]) };
    const uint32_t ob_lds[2] = { lds_off32(Ob[0]), lds_off32(Ob[1]) };

    const int q  = tid & 3;                        // 16B quarter within 32-k row
    const int r0 = tid >> 2;                       // base row for staging

    // issue one k-chunk's async copies into buffer p (8 b128 Mb + 1 b128 Ob per thread)
    auto stage = [&](int p, int kc) {
        const _Float16* g = aM + bbase + (size_t)kc + q * 8;
        const uint32_t   lb = mb_lds[p] + (uint32_t)(q * 8) * 2;
#pragma unroll
        for (int j = 0; j < 8; ++j) {
            const int row = r0 + j * 64;
            async_ld16(lb + (uint32_t)(row * MST) * 2, g + (size_t)row * D_DIM);
        }
        if (tid < 128) {
            async_ld16(ob_lds[p] + (uint32_t)(r0 * MST + q * 8) * 2,
                       aO + bbase + (size_t)(l0 + r0) * D_DIM + kc + q * 8);
        }
    };

    v8f acc[2][4];
    v8f vz = {};
#pragma unroll
    for (int rt = 0; rt < 2; ++rt)
#pragma unroll
        for (int mt = 0; mt < 4; ++mt) acc[rt][mt] = vz;

    stage(0, 0);                                   // prologue prefetch
    int p = 0;
    for (int kc = 0; kc < D_DIM; kc += 32) {
        wait_async0();                             // our copies for buf[p] landed
        __syncthreads();                           // everyone's copies visible;
                                                   // also: all reads of buf[p^1] done
        if (kc + 32 < D_DIM) stage(p ^ 1, kc + 32);// prefetch next chunk under compute

        v16h afr[2];
#pragma unroll
        for (int rt = 0; rt < 2; ++rt)
            afr[rt] = ld_frag(Ob[p] + (rt * 16 + n) * MST + kh);
#pragma unroll
        for (int mt = 0; mt < 4; ++mt) {
            v16h bfr = ld_frag(Mb[p] + (wv * 64 + mt * 16 + n) * MST + kh);
            acc[0][mt] = __builtin_amdgcn_wmma_f32_16x16x32_f16(
                false, afr[0], false, bfr, (short)0, acc[0][mt], false, false);
            acc[1][mt] = __builtin_amdgcn_wmma_f32_16x16x32_f16(
                false, afr[1], false, bfr, (short)0, acc[1][mt], false, false);
        }
        p ^= 1;
    }

    // scale scores
    const float scale = 0.05103103630798288f;      // 1/sqrt(384)
#pragma unroll
    for (int rt = 0; rt < 2; ++rt)
#pragma unroll
        for (int mt = 0; mt < 4; ++mt)
#pragma unroll
            for (int v = 0; v < 8; ++v) acc[rt][mt][v] *= scale;

    // full-row max: lane-local -> shuffle over 16 n-lanes -> cross-wave via LDS
    float pm[2][8];
#pragma unroll
    for (int rt = 0; rt < 2; ++rt)
#pragma unroll
        for (int v = 0; v < 8; ++v) {
            float m = fmaxf(fmaxf(acc[rt][0][v], acc[rt][1][v]),
                            fmaxf(acc[rt][2][v], acc[rt][3][v]));
#pragma unroll
            for (int off = 1; off < 16; off <<= 1)
                m = fmaxf(m, __shfl_xor(m, off, 32));
            pm[rt][v] = m;
        }
    if (n == 0) {
#pragma unroll
        for (int rt = 0; rt < 2; ++rt)
#pragma unroll
            for (int v = 0; v < 8; ++v)
                red[wv * 32 + rt * 16 + mb8 + v] = pm[rt][v];
    }
    __syncthreads();
    if (tid < 32) {
        float m = red[tid];
#pragma unroll
        for (int qq = 1; qq < 8; ++qq) m = fmaxf(m, red[qq * 32 + tid]);
        rowmax[tid] = m;
    }
    __syncthreads();

    // exp in-place + full-row sum
    float ps[2][8];
#pragma unroll
    for (int rt = 0; rt < 2; ++rt)
#pragma unroll
        for (int v = 0; v < 8; ++v) {
            float rm = rowmax[rt * 16 + mb8 + v];
            float s = 0.0f;
#pragma unroll
            for (int mt = 0; mt < 4; ++mt) {
                float e = __expf(acc[rt][mt][v] - rm);
                acc[rt][mt][v] = e;
                s += e;
            }
#pragma unroll
            for (int off = 1; off < 16; off <<= 1)
                s += __shfl_xor(s, off, 32);
            ps[rt][v] = s;
        }
    if (n == 0) {
#pragma unroll
        for (int rt = 0; rt < 2; ++rt)
#pragma unroll
            for (int v = 0; v < 8; ++v)
                red[wv * 32 + rt * 16 + mb8 + v] = ps[rt][v];
    }
    __syncthreads();
    if (tid < 32) {
        float s = 0.0f;
#pragma unroll
        for (int qq = 0; qq < 8; ++qq) s += red[qq * 32 + tid];
        rowinv[tid] = 1.0f / s;
    }
    __syncthreads();

    // column sums of P over this block's 32 rows -> w[b, col]
#pragma unroll
    for (int mt = 0; mt < 4; ++mt) {
        float cs = 0.0f;
#pragma unroll
        for (int rt = 0; rt < 2; ++rt)
#pragma unroll
            for (int v = 0; v < 8; ++v)
                cs += acc[rt][mt][v] * rowinv[rt * 16 + mb8 + v];
        cs += __shfl_xor(cs, 16, 32);              // merge row-halves, same column
        if (half == 0)
            atomicAdd(&wcol[b * L_DIM + wv * 64 + mt * 16 + n], cs);
    }
}

// ---------------------------------- diff[b,d] = (sum_l O - sum_m w*M) / 512

__global__ __launch_bounds__(384)
void diff_kernel(const _Float16* __restrict__ aO, const _Float16* __restrict__ aM,
                 const float* __restrict__ wcol, float* __restrict__ diff) {
    const int b = blockIdx.x, d = threadIdx.x;
    const _Float16* po = aO + (size_t)b * L_DIM * D_DIM + d;
    const _Float16* pm = aM + (size_t)b * L_DIM * D_DIM + d;
    const float* wb = wcol + b * L_DIM;
    float so = 0.0f, sw = 0.0f;
    for (int l = 0; l < L_DIM; ++l) {
        so += (float)po[(size_t)l * D_DIM];
        sw += wb[l] * (float)pm[(size_t)l * D_DIM];
    }
    diff[b * D_DIM + d] = (so - sw) * (1.0f / 512.0f);
}

// ------------------------------------------------------- tiny MLP (scalar)

__global__ __launch_bounds__(256)
void mlp_kernel(const float* __restrict__ diff,
                const float* __restrict__ W1, const float* __restrict__ b1,
                const float* __restrict__ W2, const float* __restrict__ b2,
                const float* __restrict__ W3, const float* __restrict__ b3,
                float* __restrict__ out) {
    __shared__ float df[D_DIM];
    __shared__ float x1[256];
    __shared__ float x2[64];
    const int b = blockIdx.x, t = threadIdx.x;
    for (int i = t; i < D_DIM; i += 256) df[i] = diff[b * D_DIM + i];
    __syncthreads();
    float s = b1[t];
    for (int d = 0; d < D_DIM; ++d) s += df[d] * W1[d * 256 + t];
    x1[t] = fmaxf(s, 0.0f);
    __syncthreads();
    if (t < 64) {
        float s2 = b2[t];
        for (int k = 0; k < 256; ++k) s2 += x1[k] * W2[k * 64 + t];
        x2[t] = fmaxf(s2, 0.0f);
    }
    __syncthreads();
    if (t == 0) {
        float s3 = b3[0];
        for (int k = 0; k < 64; ++k) s3 += x2[k] * W3[k];
        out[b] = s3;
    }
}

// ----------------------------------------------------------------- launcher

extern "C" void kernel_launch(void* const* d_in, const int* in_sizes, int n_in,
                              void* d_out, int out_size, void* d_ws, size_t ws_size,
                              hipStream_t stream) {
    const float* h_orig = (const float*)d_in[0];
    const float* h_mut  = (const float*)d_in[1];
    const float* gamma  = (const float*)d_in[2];
    const float* beta   = (const float*)d_in[3];
    const float* W1     = (const float*)d_in[4];
    const float* b1     = (const float*)d_in[5];
    const float* W2     = (const float*)d_in[6];
    const float* b2     = (const float*)d_in[7];
    const float* W3     = (const float*)d_in[8];
    const float* b3     = (const float*)d_in[9];
    const int* batch_o  = (const int*)d_in[10];
    const int* batch_m  = (const int*)d_in[11];
    const int* alignment= (const int*)d_in[12];
    const int* source   = (const int*)d_in[13];
    float* out = (float*)d_out;

    // workspace layout (bytes):
    //   [0,1024)        cnt_o/cnt_m/off_o/off_m (64 ints each)
    //   [1024,132096)   wcol: 64x512 f32
    //   [132096,230400) diff: 64x384 f32
    //   [230400, +24MB) aligned O (f16), then aligned M (f16)  -> ~50.6 MB total
    char* ws = (char*)d_ws;
    int* cnt_o = (int*)ws;
    int* cnt_m = cnt_o + 64;
    int* off_o = cnt_m + 64;
    int* off_m = off_o + 64;
    float* wcol = (float*)(ws + 1024);
    float* diff = (float*)(ws + 132096);
    _Float16* aO = (_Float16*)(ws + 230400);
    _Float16* aM = aO + (size_t)N_ROWS * D_DIM;

    zero_kernel<<<129, 256, 0, stream>>>((float*)ws, 33024);  // cnts + wcol
    count_kernel<<<128, 256, 0, stream>>>(batch_o, batch_m, cnt_o, cnt_m, N_ROWS);
    offsets_kernel<<<1, 1, 0, stream>>>(cnt_o, cnt_m, off_o, off_m);
    gather_ln_kernel<<<N_ROWS, 128, 0, stream>>>(h_orig, h_mut, gamma, beta,
                                                 alignment, source,
                                                 cnt_o, cnt_m, off_o, off_m, aO, aM);
    attn_kernel<<<dim3(16, 64), 256, 0, stream>>>(aO, aM, wcol);
    diff_kernel<<<64, 384, 0, stream>>>(aO, aM, wcol, diff);
    mlp_kernel<<<64, 256, 0, stream>>>(diff, W1, b1, W2, b2, W3, b3, out);
}
